// CrossLayer_25220047962582
// MI455X (gfx1250) — matrix operand order repaired and checked by reading, not compiled, not run on hardware
//
#include <hip/hip_runtime.h>

#define NPTS 8192
#define KNN  16
#define CFEAT 64
#define C_IN 131
#define C_PAD 160          // 131 padded to a multiple of 32 (5 K-steps of 32)
#define LEAKY 0.1f
#define QPB 64             // queries per block (8 waves x 8 iterations)

typedef __attribute__((ext_vector_type(16))) _Float16 v16h;
typedef __attribute__((ext_vector_type(8)))  _Float16 v8h;
typedef __attribute__((ext_vector_type(8)))  float    v8f;

// ---------------------------------------------------------------------------
// Stage 1: brute-force 16-NN. One wave32 per query point.
// ---------------------------------------------------------------------------
__global__ __launch_bounds__(256) void knn_kernel(
    const float* __restrict__ qxyz,   // (B,3,N)
    const float* __restrict__ cxyz,   // (B,3,N)
    int* __restrict__ out_idx)        // (B,N,KNN)
{
  __shared__ float s_d[8][32 * KNN];
  __shared__ int   s_i[8][32 * KNN];

  const int lane = threadIdx.x & 31;
  const int wave = threadIdx.x >> 5;
  const int gq = blockIdx.x * 8 + wave;
  const int b = gq / NPTS;
  const int q = gq - b * NPTS;

  const float* qx = qxyz + (size_t)b * 3 * NPTS;
  const float* cx = cxyz + (size_t)b * 3 * NPTS;
  const float ax = qx[q], ay = qx[NPTS + q], az = qx[2 * NPTS + q];

  float dist[KNN]; int idx[KNN];
#pragma unroll
  for (int j = 0; j < KNN; ++j) { dist[j] = 3.4e38f; idx[j] = 0x7fffffff; }

  // lane-strided scan over all candidates, per-lane sorted top-16
  for (int i = lane; i < NPTS; i += 32) {
    const float dx = ax - cx[i];
    const float dy = ay - cx[NPTS + i];
    const float dz = az - cx[2 * NPTS + i];
    const float d = dx * dx + dy * dy + dz * dz;
    if (d < dist[KNN - 1]) {
      float dj = d; int ij = i;
#pragma unroll
      for (int j = 0; j < KNN; ++j) {
        const float td = dist[j]; const int ti = idx[j];
        const bool sw = (dj < td) || (dj == td && ij < ti);
        dist[j] = sw ? dj : td;  idx[j] = sw ? ij : ti;
        dj      = sw ? td : dj;  ij     = sw ? ti : ij;
      }
    }
  }

  // spill sorted lists to LDS, then 16-round cross-lane merge
  float* sd = s_d[wave];
  int*   si = s_i[wave];
#pragma unroll
  for (int j = 0; j < KNN; ++j) { sd[lane * KNN + j] = dist[j]; si[lane * KNN + j] = idx[j]; }
  asm volatile("s_wait_dscnt 0" ::: "memory");

  int p = 0;
  for (int r = 0; r < KNN; ++r) {
    float v  = sd[lane * KNN + p];
    int   id = si[lane * KNN + p];
    int   ln = lane;
#pragma unroll
    for (int off = 16; off >= 1; off >>= 1) {
      const float ov = __shfl_xor(v, off, 32);
      const int  oid = __shfl_xor(id, off, 32);
      const int  oln = __shfl_xor(ln, off, 32);
      if (ov < v || (ov == v && oid < id)) { v = ov; id = oid; ln = oln; }
    }
    if (lane == ln) ++p;
    if (lane == 0) out_idx[((size_t)b * NPTS + q) * KNN + r] = id;
  }
}

// ---------------------------------------------------------------------------
// WMMA fragment loaders matching the CDNA5 16-bit A/B VGPR layouts.
// ---------------------------------------------------------------------------
__device__ __forceinline__ v16h load_a_frag(const _Float16* X, int xstride, int ks, int lane) {
  // A (16x32, MxK): lane<16 row=lane holds K {0..7,16..23}; lane>=16 holds {8..15,24..31}
  const int row = lane & 15;
  const int kb  = ks * 32 + ((lane >> 4) << 3);
  const _Float16* p = X + row * xstride + kb;
  const v8h lo = *(const v8h*)(p);
  const v8h hi = *(const v8h*)(p + 16);
  v16h a;
#pragma unroll
  for (int t = 0; t < 8; ++t) { a[t] = lo[t]; a[t + 8] = hi[t]; }
  return a;
}

__device__ __forceinline__ v16h load_b_frag(const _Float16* W, int wstride, int ks, int nt, int lane) {
  // B (32x16, KxN): lane<16 col=lane holds K {0..7,16..23}; lane>=16 holds {8..15,24..31}
  const int col = nt * 16 + (lane & 15);
  const int kb  = ks * 32 + ((lane >> 4) << 3);
  const _Float16* p = W + col * wstride + kb;
  const v8h lo = *(const v8h*)(p);
  const v8h hi = *(const v8h*)(p + 16);
  v16h b;
#pragma unroll
  for (int t = 0; t < 8; ++t) { b[t] = lo[t]; b[t + 8] = hi[t]; }
  return b;
}

__device__ __forceinline__ void mlp_layer(const _Float16* X, int xstride, int ksteps,
                                          const _Float16* W, int wstride,
                                          int lane, v8f acc[4]) {
#pragma unroll
  for (int t = 0; t < 4; ++t) acc[t] = (v8f){};
  for (int ks = 0; ks < ksteps; ++ks) {
    const v16h a = load_a_frag(X, xstride, ks, lane);
#pragma unroll
    for (int t = 0; t < 4; ++t) {
      const v16h bm = load_b_frag(W, wstride, ks, t, lane);
      acc[t] = __builtin_amdgcn_wmma_f32_16x16x32_f16(
          false, a, false, bm, (short)0, acc[t], false, false);
    }
  }
}

// ---------------------------------------------------------------------------
// Stage 2: fused gather + MLP (WMMA) + max-pool over K neighbors.
// One wave per query; D-tile = (query's 16 neighbors) x (16 channels).
// ---------------------------------------------------------------------------
__global__ __launch_bounds__(256) void cross_mlp_kernel(
    const float* __restrict__ qxyz,  const float* __restrict__ nxyz,   // (B,3,N)
    const float* __restrict__ qfeat, const float* __restrict__ nfeat,  // (B,64,N)
    const int*   __restrict__ knn_idx,                                  // (B,N,16)
    const float* __restrict__ w0p, const float* __restrict__ bias0,     // (64,131),(64)
    const float* __restrict__ w1p, const float* __restrict__ bias1,     // (64,64),(64)
    const float* __restrict__ w2p, const float* __restrict__ bias2,     // (64,64),(64) or null
    float* __restrict__ out, int nlayers)                               // (B,64,N)
{
  extern __shared__ __align__(64) char smem_raw[];
  _Float16* sW0 = (_Float16*)smem_raw;       // 64*160 f16 (zero-padded cols)
  _Float16* sW1 = sW0 + 64 * C_PAD;          // 64*64
  _Float16* sW2 = sW1 + 64 * 64;             // 64*64
  float*    sB  = (float*)(sW2 + 64 * 64);   // 3*64
  _Float16* sXall = (_Float16*)(sB + 3 * 64);

  const int tid  = threadIdx.x;
  const int lane = tid & 31;
  const int wave = tid >> 5;
  _Float16* sX = sXall + wave * (16 * C_PAD);   // per-wave activation tile

  // cooperative weight/bias staging (fp32 -> fp16)
  for (int i = tid; i < 64 * C_PAD; i += 256) {
    const int r = i / C_PAD, c = i - r * C_PAD;
    sW0[i] = (c < C_IN) ? (_Float16)w0p[r * C_IN + c] : (_Float16)0.f;
  }
  for (int i = tid; i < 64 * 64; i += 256) sW1[i] = (_Float16)w1p[i];
  if (nlayers == 3)
    for (int i = tid; i < 64 * 64; i += 256) sW2[i] = (_Float16)w2p[i];
  if (tid < 64) {
    sB[tid]       = bias0[tid];
    sB[64 + tid]  = bias1[tid];
    sB[128 + tid] = (nlayers == 3) ? bias2[tid] : 0.f;
  }
  __syncthreads();

  const int qbase = blockIdx.x * QPB;
  for (int qi = wave; qi < QPB; qi += 8) {
    const int gq = qbase + qi;
    const int b = gq / NPTS;
    const int q = gq - b * NPTS;

    const float* qf = qfeat + (size_t)b * CFEAT * NPTS;
    const float* nf = nfeat + (size_t)b * CFEAT * NPTS;
    const float* qx = qxyz  + (size_t)b * 3 * NPTS;
    const float* nx = nxyz  + (size_t)b * 3 * NPTS;
    const int*   nb = knn_idx + ((size_t)b * NPTS + q) * KNN;

    const int row = lane & 15;
    const int ii  = nb[row];    // neighbor index for this row

    // -- build X = [qfeat | nfeat(nbr) | direction | 0-pad], 16 x 160 f16 --
    {
      const _Float16 h0 = (_Float16)qf[lane * NPTS + q];
      const _Float16 h1 = (_Float16)qf[(lane + 32) * NPTS + q];
#pragma unroll
      for (int r = 0; r < 16; ++r) {
        sX[r * C_PAD + lane]      = h0;
        sX[r * C_PAD + lane + 32] = h1;
      }
    }
    {
      const int co = lane >> 4;          // channel parity handled by lane half
#pragma unroll
      for (int c = 0; c < CFEAT; c += 2)
        sX[row * C_PAD + CFEAT + c + co] = (_Float16)nf[(c + co) * NPTS + ii];
    }
    if (lane < 16) {
      const float ax = qx[q], ay = qx[NPTS + q], az = qx[2 * NPTS + q];
      sX[row * C_PAD + 128] = (_Float16)(nx[ii] - ax);
      sX[row * C_PAD + 129] = (_Float16)(nx[NPTS + ii] - ay);
      sX[row * C_PAD + 130] = (_Float16)(nx[2 * NPTS + ii] - az);
#pragma unroll
      for (int c = C_IN; c < C_PAD; ++c) sX[row * C_PAD + c] = (_Float16)0.f;
    }
    asm volatile("s_wait_dscnt 0" ::: "memory");

    v8f acc[4];
    const int mofs = (lane >> 4) * 8;   // D layout: lane half selects rows 0-7 / 8-15
    const int n_lo = lane & 15;

    // ---- layer 0: (16x160) x (160x64), 5 K-steps x 4 N-tiles ----
    mlp_layer(sX, C_PAD, C_PAD / 32, sW0, C_PAD, lane, acc);
#pragma unroll
    for (int t = 0; t < 4; ++t) {
      const float bias = sB[t * 16 + n_lo];
#pragma unroll
      for (int v = 0; v < 8; ++v) {
        float x = acc[t][v] + bias;
        x = (x > 0.f) ? x : LEAKY * x;
        sX[(mofs + v) * 64 + t * 16 + n_lo] = (_Float16)x;   // stride-64 activations
      }
    }
    asm volatile("s_wait_dscnt 0" ::: "memory");

    // ---- layer 1: (16x64) x (64x64) ----
    mlp_layer(sX, 64, 2, sW1, 64, lane, acc);
    if (nlayers == 3) {
#pragma unroll
      for (int t = 0; t < 4; ++t) {
        const float bias = sB[64 + t * 16 + n_lo];
#pragma unroll
        for (int v = 0; v < 8; ++v) {
          float x = acc[t][v] + bias;
          x = (x > 0.f) ? x : LEAKY * x;
          sX[(mofs + v) * 64 + t * 16 + n_lo] = (_Float16)x;
        }
      }
      asm volatile("s_wait_dscnt 0" ::: "memory");
      // ---- layer 2 ----
      mlp_layer(sX, 64, 2, sW2, 64, lane, acc);
    }

    // ---- max-pool over 16 neighbor rows, then bias + leaky (monotonic) ----
    const float* bl = sB + (nlayers == 3 ? 128 : 64);
#pragma unroll
    for (int t = 0; t < 4; ++t) {
      float m = acc[t][0];
#pragma unroll
      for (int v = 1; v < 8; ++v) m = fmaxf(m, acc[t][v]);   // 8 rows in-lane
      m = fmaxf(m, __shfl_xor(m, 16, 32));                   // merge row halves
      float x = m + bl[t * 16 + n_lo];
      x = (x > 0.f) ? x : LEAKY * x;
      if (lane < 16)
        out[((size_t)b * CFEAT + t * 16 + n_lo) * NPTS + q] = x;
    }
  }
}

// ---------------------------------------------------------------------------
extern "C" void kernel_launch(void* const* d_in, const int* in_sizes, int n_in,
                              void* d_out, int out_size, void* d_ws, size_t ws_size,
                              hipStream_t stream)
{
  const float* pc1   = (const float*)d_in[0];
  const float* pc2   = (const float*)d_in[1];
  const float* feat1 = (const float*)d_in[2];
  const float* feat2 = (const float*)d_in[3];
  const float* w1_0 = (const float*)d_in[4];  const float* b1_0 = (const float*)d_in[5];
  const float* w1_1 = (const float*)d_in[6];  const float* b1_1 = (const float*)d_in[7];
  const float* w1_2 = (const float*)d_in[8];  const float* b1_2 = (const float*)d_in[9];
  const float* w2_0 = (const float*)d_in[10]; const float* b2_0 = (const float*)d_in[11];
  const float* w2_1 = (const float*)d_in[12]; const float* b2_1 = (const float*)d_in[13];

  const int B  = in_sizes[0] / (3 * NPTS);
  const int nq = B * NPTS;

  int* idx12 = (int*)d_ws;
  int* idx21 = idx12 + (size_t)nq * KNN;

  float* out_f1new = (float*)d_out;
  float* out_f2new = out_f1new + (size_t)B * CFEAT * NPTS;
  float* out_f1fin = out_f2new + (size_t)B * CFEAT * NPTS;

  const dim3 blk(256);

  knn_kernel<<<nq / 8, blk, 0, stream>>>(pc1, pc2, idx12);
  knn_kernel<<<nq / 8, blk, 0, stream>>>(pc2, pc1, idx21);

  const size_t smem = (size_t)(64 * C_PAD + 2 * 64 * 64) * 2   // f16 weights
                    + 3 * 64 * 4                               // biases
                    + (size_t)8 * 16 * C_PAD * 2;              // per-wave X tiles

  cross_mlp_kernel<<<nq / QPB, blk, smem, stream>>>(
      pc1, pc2, feat1, feat2, idx12,
      w1_0, b1_0, w1_1, b1_1, w1_2, b1_2, out_f1new, 3);
  cross_mlp_kernel<<<nq / QPB, blk, smem, stream>>>(
      pc2, pc1, feat2, feat1, idx21,
      w1_0, b1_0, w1_1, b1_1, w1_2, b1_2, out_f2new, 3);
  cross_mlp_kernel<<<nq / QPB, blk, smem, stream>>>(
      pc1, pc2, out_f1new, out_f2new, idx12,
      w2_0, b2_0, w2_1, b2_1, nullptr, nullptr, out_f1fin, 2);
}
